// SpeechEmbedder_27487790694457
// MI455X (gfx1250) — matrix operand order repaired and checked by reading, not compiled, not run on hardware
//
#include <hip/hip_runtime.h>
#include <hip/hip_bf16.h>

typedef _Float16 h16;
typedef __attribute__((ext_vector_type(16))) _Float16 v16h;
typedef __attribute__((ext_vector_type(8)))  _Float16 v8h;
typedef __attribute__((ext_vector_type(8)))  float    v8f;

union V16 { v16h v; v8h h[2]; };

#define BB 320
#define TT 120
#define HHID 768
#define DDIM 256

// ---------- prep kernels ----------

__global__ void cvt_pad_kernel(const float* __restrict__ src, h16* __restrict__ dst,
                               int rows, int scols, int dcols) {
  int i = blockIdx.x * blockDim.x + threadIdx.x;
  if (i >= rows * dcols) return;
  int r = i / dcols, c = i - r * dcols;
  dst[i] = (c < scols) ? (h16)src[(size_t)r * scols + c] : (h16)0.f;
}

__global__ void bias_kernel(const float* __restrict__ a, const float* __restrict__ b,
                            float* __restrict__ o, int n) {
  int i = blockIdx.x * blockDim.x + threadIdx.x;
  if (i < n) o[i] = a[i] + b[i];
}

// mel (320, 40, 120) -> x0 (320, 120, 64) f16, zero-padded K 40->64
__global__ void mel_kernel(const float* __restrict__ mel, h16* __restrict__ x0) {
  int i = blockIdx.x * blockDim.x + threadIdx.x;
  if (i >= BB * TT * 64) return;
  int kk = i & 63;
  int bt = i >> 6;
  int t = bt % TT;
  int b = bt / TT;
  x0[i] = (kk < 40) ? (h16)mel[((size_t)b * 40 + kk) * TT + t] : (h16)0.f;
}

// ---------- software-pipelined GEMM phase for the LSTM step ----------
// acc[i] += A(rows of arows[i], K) @ B(W row n, K)^T over ksteps K-chunks of 32.
// A fragment for the next wmma (and B for the next K-step) is loaded BEFORE the
// current wmma issues, in distinct registers, so the XDL op hides load latency.

__device__ __forceinline__ v16h load_bfrag(const h16* wrow, int k0, int hh) {
  V16 b;
  b.h[0] = *(const v8h*)(wrow + k0 + hh * 8);
  b.h[1] = *(const v8h*)(wrow + k0 + 16 + hh * 8);
  return b.v;
}

__device__ __forceinline__ V16 load_afrag(const h16* arow, int k0, int hh) {
  V16 a;
  a.h[0] = *(const v8h*)(arow + k0 + hh * 8);
  a.h[1] = *(const v8h*)(arow + k0 + 16 + hh * 8);
  return a;
}

__device__ __forceinline__ void gemm_phase(v8f (&acc)[5], const h16* wrow,
                                           const h16* const (&arows)[5],
                                           int ksteps, int hh) {
  v16h bc = load_bfrag(wrow, 0, hh);
  V16 ac = load_afrag(arows[0], 0, hh);
  for (int ks = 0; ks < ksteps - 1; ++ks) {
    const int k0 = ks * 32;
#pragma unroll
    for (int i = 0; i < 5; ++i) {
      V16 an;
      v16h bn;
      if (i < 4) {
        an = load_afrag(arows[i + 1], k0, hh);
      } else {
        bn = load_bfrag(wrow, k0 + 32, hh);
        an = load_afrag(arows[0], k0 + 32, hh);
      }
      acc[i] = __builtin_amdgcn_wmma_f32_16x16x32_f16(
          false, ac.v, false, bc, (short)0, acc[i], false, false);
      ac = an;
      if (i == 4) bc = bn;
    }
  }
  {  // last K-step (no B prefetch)
    const int k0 = (ksteps - 1) * 32;
#pragma unroll
    for (int i = 0; i < 5; ++i) {
      V16 an;
      if (i < 4) an = load_afrag(arows[i + 1], k0, hh);
      acc[i] = __builtin_amdgcn_wmma_f32_16x16x32_f16(
          false, ac.v, false, bc, (short)0, acc[i], false, false);
      if (i < 4) ac = an;
    }
  }
}

// ---------- fused LSTM timestep: gates GEMM (WMMA) + cell update ----------
// grid: 96 blocks = 4 row-quarters (80 rows) x 24 col-strips (32 cols of H),
// 256 threads (8 waves = 4 gates x 2 16-col subtiles).
// Each wave: one (gate, 16-col) strip; per K-step it loads the weight (B)
// fragment ONCE and reuses it across 5 M-tiles (weights are the large,
// L2-resident operand; h/x reuse is absorbed by WGP$). After the GEMM the
// block holds all 4 gate tiles for its (rows, j) region; they are exchanged
// through LDS and the LSTM cell update is applied in the same kernel.

__global__ void __launch_bounds__(256)
lstm_step_kernel(const h16* __restrict__ xin, int ld_in, int kx_steps,
                 const h16* __restrict__ Wih, int ldWih,
                 const h16* __restrict__ Whh,
                 const float* __restrict__ bias,
                 const h16* __restrict__ h_in,
                 float* __restrict__ cbuf,
                 h16* __restrict__ h_out,
                 h16* __restrict__ xout,
                 int t) {
  __shared__ float lds[4 * 80 * 32];   // 40 KB: 4 gates x 80 rows x 32 cols

  const int mq = blockIdx.x / 24;      // 0..3  -> rows m0..m0+79
  const int cb = blockIdx.x % 24;      // 0..23 -> cols j0..j0+31 (of H)
  const int m0 = mq * 80;
  const int j0 = cb * 32;
  const int wave = threadIdx.x >> 5;   // 0..7
  const int lane = threadIdx.x & 31;
  const int gate = wave >> 1;          // 0..3 = i,f,g,o
  const int nsub = wave & 1;           // which 16-col subtile
  const int l15 = lane & 15;
  const int hh = lane >> 4;
  const int n = gate * HHID + j0 + nsub * 16 + l15;  // absolute gate column

  v8f acc[5];
  {
    float bv = bias[n];
#pragma unroll
    for (int i = 0; i < 5; ++i)
#pragma unroll
      for (int r = 0; r < 8; ++r) acc[i][r] = bv;
  }

  // Phase X: gates += x_t @ Wih^T
  {
    const h16* arows[5];
#pragma unroll
    for (int i = 0; i < 5; ++i)
      arows[i] = xin + ((size_t)(m0 + i * 16 + l15) * TT + t) * ld_in;
    gemm_phase(acc, Wih + (size_t)n * ldWih, arows, kx_steps, hh);
  }

  // Phase H: gates += h_{t-1} @ Whh^T  (h_0 == 0 -> skip)
  if (t > 0) {
    const h16* arows[5];
#pragma unroll
    for (int i = 0; i < 5; ++i)
      arows[i] = h_in + (size_t)(m0 + i * 16 + l15) * HHID;
    gemm_phase(acc, Whh + (size_t)n * HHID, arows, 24, hh);
  }

  // spill gate tiles to LDS (C layout: lane l, reg r -> M=(l>>4)*8+r, N=l&15)
#pragma unroll
  for (int i = 0; i < 5; ++i)
#pragma unroll
    for (int r = 0; r < 8; ++r) {
      int mloc = i * 16 + hh * 8 + r;        // 0..79
      int nloc = nsub * 16 + l15;            // 0..31
      lds[(gate * 80 + mloc) * 32 + nloc] = acc[i][r];
    }
  __syncthreads();

  // pointwise LSTM cell: 80x32 = 2560 elems over 256 threads
  for (int it = 0; it < 10; ++it) {
    int idx = threadIdx.x + it * 256;
    int row = idx >> 5;
    int col = idx & 31;
    float gi = lds[(0 * 80 + row) * 32 + col];
    float gf = lds[(1 * 80 + row) * 32 + col];
    float gg = lds[(2 * 80 + row) * 32 + col];
    float go = lds[(3 * 80 + row) * 32 + col];
    int m = m0 + row, j = j0 + col;
    float c_old = (t > 0) ? cbuf[(size_t)m * HHID + j] : 0.f;
    float ig = 1.f / (1.f + __expf(-gi));
    float fg = 1.f / (1.f + __expf(-gf));
    float og = 1.f / (1.f + __expf(-go));
    float cn = fg * c_old + ig * tanhf(gg);
    float hn = og * tanhf(cn);
    cbuf[(size_t)m * HHID + j] = cn;
    h_out[(size_t)m * HHID + j] = (h16)hn;
    xout[((size_t)m * TT + t) * HHID + j] = (h16)hn;
  }
}

// ---------- projection GEMM (WMMA): dvec_raw = h_last @ Wproj^T + bproj ----------
// 320x256, K=768; one 16x16 tile per wave; 40 blocks x 8 waves.
__global__ void __launch_bounds__(256)
proj_kernel(const h16* __restrict__ xlast_base, const h16* __restrict__ Wp,
            const float* __restrict__ bp, float* __restrict__ dvec) {
  const int wave = threadIdx.x >> 5, lane = threadIdx.x & 31;
  const int w = blockIdx.x * 8 + wave;   // 0..319
  const int mt = w >> 4, nt = w & 15;
  const int l15 = lane & 15, hh = lane >> 4;
  const int mrow = mt * 16 + l15;
  const int n = nt * 16 + l15;

  v8f acc;
  float bv = bp[n];
#pragma unroll
  for (int r = 0; r < 8; ++r) acc[r] = bv;

  const h16* arow = xlast_base + ((size_t)mrow * TT + (TT - 1)) * HHID;
  const h16* brow = Wp + (size_t)n * HHID;

  V16 ac = load_afrag(arow, 0, hh);
  v16h bc = load_bfrag(brow, 0, hh);
  for (int ks = 0; ks < 23; ++ks) {
    const int k0 = ks * 32;
    V16 an = load_afrag(arow, k0 + 32, hh);
    v16h bn = load_bfrag(brow, k0 + 32, hh);
    acc = __builtin_amdgcn_wmma_f32_16x16x32_f16(
        false, ac.v, false, bc, (short)0, acc, false, false);
    ac = an;
    bc = bn;
  }
  acc = __builtin_amdgcn_wmma_f32_16x16x32_f16(
      false, ac.v, false, bc, (short)0, acc, false, false);

#pragma unroll
  for (int r = 0; r < 8; ++r)
    dvec[(size_t)(mt * 16 + hh * 8 + r) * DDIM + n] = acc[r];
}

// ---------- L2 normalize rows of dvec (in place), one block per row ----------
__global__ void __launch_bounds__(256)
normalize_kernel(float* __restrict__ dvec) {
  __shared__ float sm[256];
  int b = blockIdx.x, tid = threadIdx.x;
  float v = dvec[(size_t)b * DDIM + tid];
  sm[tid] = v * v;
  __syncthreads();
  for (int s = 128; s > 0; s >>= 1) {
    if (tid < s) sm[tid] += sm[tid + s];
    __syncthreads();
  }
  dvec[(size_t)b * DDIM + tid] = v / sqrtf(sm[0]);
}

// ---------- centroids: cent[s][d] = mean_u dvec[s*U+u][d] ----------
__global__ void __launch_bounds__(256)
cent_kernel(const float* __restrict__ dvec, float* __restrict__ cent) {
  int s = blockIdx.x, d = threadIdx.x;
  float acc = 0.f;
  for (int u = 0; u < 10; ++u) acc += dvec[((size_t)s * 10 + u) * DDIM + d];
  cent[(size_t)s * DDIM + d] = acc * 0.1f;
}

// ---------- GE2E loss: one block, 320 threads (one per (s,u)) ----------
__global__ void __launch_bounds__(320)
ge2e_kernel(const float* __restrict__ dvec, const float* __restrict__ cent,
            const float* __restrict__ lw, const float* __restrict__ lb,
            float* __restrict__ out) {
  __shared__ float ncs[32];
  __shared__ float rl[320], rp[320], rn[320];
  int tid = threadIdx.x;
  if (tid < 32) {
    float s2 = 0.f;
    for (int d = 0; d < DDIM; ++d) { float v = cent[tid * DDIM + d]; s2 += v * v; }
    ncs[tid] = sqrtf(s2);
  }
  __syncthreads();

  int s = tid / 10;
  const float* d  = dvec + (size_t)tid * DDIM;
  const float* cs = cent + (size_t)s * DDIM;
  float dd = 0.f, dpc = 0.f, pcpc = 0.f;
  for (int k = 0; k < DDIM; ++k) {
    float dv = d[k], cv = cs[k];
    float pc = (cv * 10.f - dv) * (1.f / 9.f);
    dd += dv * dv; dpc += dv * pc; pcpc += pc * pc;
  }
  float w = lw[0], b = lb[0];
  float na  = fmaxf(sqrtf(dd),   1e-6f);
  float npc = fmaxf(sqrtf(pcpc), 1e-6f);
  float pos_sim = w * (dpc / (na * npc)) + b;
  float nd = fmaxf(sqrtf(dd), 1e-8f);

  float sims[31];
  int cnt = 0;
  float mx = -1e30f, negsum = 0.f;
  for (int k = 0; k < 32; ++k) {
    if (k == s) continue;
    const float* ck = cent + (size_t)k * DDIM;
    float dot = 0.f;
    for (int q = 0; q < DDIM; ++q) dot += d[q] * ck[q];
    float nc = fmaxf(ncs[k], 1e-8f);
    float sim = w * (dot / (nd * nc)) + b;
    sims[cnt++] = sim;
    mx = fmaxf(mx, sim);
    negsum += sim;
  }
  float se = 0.f;
  for (int k = 0; k < 31; ++k) se += __expf(sims[k] - mx);
  float lse = mx + __logf(se);
  float lossv = -pos_sim + lse;

  rl[tid] = lossv; rp[tid] = pos_sim; rn[tid] = negsum;
  __syncthreads();
  if (tid == 0) {
    float L = 0.f, P = 0.f, N = 0.f;
    for (int i = 0; i < 320; ++i) { L += rl[i]; P += rp[i]; N += rn[i]; }
    out[0] = L / 320.f;
    out[1] = P / 320.f;
    out[2] = N / (320.f * 31.f);
  }
}

// ---------- host ----------

extern "C" void kernel_launch(void* const* d_in, const int* in_sizes, int n_in,
                              void* d_out, int out_size, void* d_ws, size_t ws_size,
                              hipStream_t stream) {
  const float* mel   = (const float*)d_in[0];
  const float* Wih[3] = {(const float*)d_in[1], (const float*)d_in[5], (const float*)d_in[9]};
  const float* Whh[3] = {(const float*)d_in[2], (const float*)d_in[6], (const float*)d_in[10]};
  const float* bih[3] = {(const float*)d_in[3], (const float*)d_in[7], (const float*)d_in[11]};
  const float* bhh[3] = {(const float*)d_in[4], (const float*)d_in[8], (const float*)d_in[12]};
  const float* Wproj = (const float*)d_in[13];
  const float* bproj = (const float*)d_in[14];
  const float* lw    = (const float*)d_in[15];
  const float* lb    = (const float*)d_in[16];
  float* out = (float*)d_out;

  char* ws = (char*)d_ws;
  size_t off = 0;
  auto alloc = [&](size_t bytes) -> char* {
    char* p = ws + off;
    off += (bytes + 255) & ~(size_t)255;
    return p;
  };

  const size_t G4H = 4 * (size_t)HHID;                 // 3072
  h16*  Wih16[3];
  h16*  Whh16[3];
  float* biasc[3];
  Wih16[0] = (h16*)alloc(G4H * 64 * 2);
  Whh16[0] = (h16*)alloc(G4H * HHID * 2);
  Wih16[1] = (h16*)alloc(G4H * HHID * 2);
  Whh16[1] = (h16*)alloc(G4H * HHID * 2);
  Wih16[2] = (h16*)alloc(G4H * HHID * 2);
  Whh16[2] = (h16*)alloc(G4H * HHID * 2);
  h16* Wp16 = (h16*)alloc((size_t)DDIM * HHID * 2);
  biasc[0] = (float*)alloc(G4H * 4);
  biasc[1] = (float*)alloc(G4H * 4);
  biasc[2] = (float*)alloc(G4H * 4);
  float* cbuf = (float*)alloc((size_t)BB * HHID * 4);
  h16* hA = (h16*)alloc((size_t)BB * HHID * 2);
  h16* hB = (h16*)alloc((size_t)BB * HHID * 2);
  h16* x0    = (h16*)alloc((size_t)BB * TT * 64 * 2);
  h16* xbufA = (h16*)alloc((size_t)BB * TT * HHID * 2);
  h16* xbufB = (h16*)alloc((size_t)BB * TT * HHID * 2);
  float* dvec = (float*)alloc((size_t)BB * DDIM * 4);
  float* cent = (float*)alloc((size_t)32 * DDIM * 4);
  (void)ws_size; (void)in_sizes; (void)n_in; (void)out_size;

  auto cdiv = [](int a, int b) { return (a + b - 1) / b; };

  // weight conversion to f16 (padded K for layer 0)
  cvt_pad_kernel<<<cdiv(3072 * 64, 256), 256, 0, stream>>>(Wih[0], Wih16[0], 3072, 40, 64);
  cvt_pad_kernel<<<cdiv(3072 * 768, 256), 256, 0, stream>>>(Whh[0], Whh16[0], 3072, 768, 768);
  cvt_pad_kernel<<<cdiv(3072 * 768, 256), 256, 0, stream>>>(Wih[1], Wih16[1], 3072, 768, 768);
  cvt_pad_kernel<<<cdiv(3072 * 768, 256), 256, 0, stream>>>(Whh[1], Whh16[1], 3072, 768, 768);
  cvt_pad_kernel<<<cdiv(3072 * 768, 256), 256, 0, stream>>>(Wih[2], Wih16[2], 3072, 768, 768);
  cvt_pad_kernel<<<cdiv(3072 * 768, 256), 256, 0, stream>>>(Whh[2], Whh16[2], 3072, 768, 768);
  cvt_pad_kernel<<<cdiv(256 * 768, 256), 256, 0, stream>>>(Wproj, Wp16, 256, 768, 768);
  for (int l = 0; l < 3; ++l)
    bias_kernel<<<cdiv(3072, 256), 256, 0, stream>>>(bih[l], bhh[l], biasc[l], 3072);

  // mel -> (B, T, 64) f16
  mel_kernel<<<cdiv(BB * TT * 64, 256), 256, 0, stream>>>(mel, x0);

  // three LSTM layers, 120 timesteps each
  for (int l = 0; l < 3; ++l) {
    const h16* xin = (l == 0) ? x0 : ((l == 1) ? xbufA : xbufB);
    h16* xout      = (l == 0) ? xbufA : ((l == 1) ? xbufB : xbufA);
    int ld_in = (l == 0) ? 64 : HHID;
    int kx    = (l == 0) ? 2 : 24;
    for (int t = 0; t < TT; ++t) {
      h16* hout      = (t & 1) ? hB : hA;
      const h16* hin = (t & 1) ? hA : hB;
      lstm_step_kernel<<<96, 256, 0, stream>>>(
          xin, ld_in, kx, Wih16[l], ld_in, Whh16[l], biasc[l],
          hin, cbuf, hout, xout, t);
    }
  }

  // projection, normalize, centroids, GE2E (layer-2 output is in xbufA)
  proj_kernel<<<40, 256, 0, stream>>>(xbufA, Wp16, bproj, dvec);
  normalize_kernel<<<BB, 256, 0, stream>>>(dvec);
  cent_kernel<<<32, 256, 0, stream>>>(dvec, cent);
  ge2e_kernel<<<1, 320, 0, stream>>>(dvec, cent, lw, lb, out);
}
